// MultiHeadAttention_80376017977806
// MI455X (gfx1250) — matrix operand order repaired and checked by reading, not compiled, not run on hardware
//
#include <hip/hip_runtime.h>
#include <hip/hip_bf16.h>
#include <math.h>

#define B_  8
#define S_  1024
#define D_  1024
#define H_  16
#define DH_ 64
#define NT  (B_ * S_)    // 8192 tokens
#define HD  (H_ * DH_)   // 1024

typedef __attribute__((ext_vector_type(16))) __bf16 v16bf;
typedef __attribute__((ext_vector_type(8)))  __bf16 v8bf;
typedef __attribute__((ext_vector_type(8)))  float  v8f;

static __device__ __forceinline__ v16bf cat8(v8bf lo, v8bf hi) {
  v16bf r;
#pragma unroll
  for (int i = 0; i < 8; ++i) { r[i] = lo[i]; r[i + 8] = hi[i]; }
  return r;
}

static __device__ __forceinline__ v8f wmma_bf16(v16bf a, v16bf b, v8f c) {
  return __builtin_amdgcn_wmma_f32_16x16x32_bf16(false, a, false, b, (short)0, c,
                                                 false, false);
}

// pack two floats -> two bf16 in one dword (lo at lower address)
static __device__ __forceinline__ unsigned pack2f(float a, float b) {
  const __bf16 lo = (__bf16)a, hi = (__bf16)b;
  return ((unsigned)__builtin_bit_cast(unsigned short, hi) << 16) |
         (unsigned)__builtin_bit_cast(unsigned short, lo);
}
static __device__ __forceinline__ unsigned pack2b(__bf16 a, __bf16 b) {
  return ((unsigned)__builtin_bit_cast(unsigned short, b) << 16) |
         (unsigned)__builtin_bit_cast(unsigned short, a);
}

// ---------------------------------------------------------------------------
// GEMM: Out[b,h,s,dh](bf16) = X(fp32, NTxD) @ W(fp32, DxHD) + bias
// 128 threads (4 waves), tile 64x64, K-step 64, 8 WMMA / wave / iter.
// ---------------------------------------------------------------------------
__global__ __launch_bounds__(128)
void qkv_proj_kernel(const float* __restrict__ X, const float* __restrict__ W,
                     const float* __restrict__ bias, __bf16* __restrict__ Out) {
  __shared__ __align__(32) __bf16 As[64][64];   // A tile, row-major (m, k)
  __shared__ __align__(32) __bf16 Bt[64][64];   // B tile, transposed (n, k)

  const int t    = threadIdx.x;
  const int wv   = t >> 5;
  const int lane = t & 31;
  const int half = lane >> 4;
  const int l16  = lane & 15;
  const int wr   = wv >> 1, wc = wv & 1;

  const int m0 = (blockIdx.x >> 4) * 64;  // 128 M tiles
  const int n0 = (blockIdx.x & 15) * 64;  // 16 N tiles

  v8f acc[2][2] = {{{}, {}}, {{}, {}}};

  for (int k0 = 0; k0 < D_; k0 += 64) {
    __syncthreads();
    // stage A: X[m0..+63][k0..+63] fp32 -> bf16, packed ds_store_b64
#pragma unroll
    for (int i = 0; i < 8; ++i) {
      const int f = t * 8 + i;                 // 1024 float4 total
      const int row = f >> 4, c4 = f & 15;
      const float4 v = *(const float4*)(X + (size_t)(m0 + row) * D_ + k0 + c4 * 4);
      uint2 u;
      u.x = pack2f(v.x, v.y);
      u.y = pack2f(v.z, v.w);
      *(uint2*)&As[row][c4 * 4] = u;
    }
    // stage B transposed: W[k0..+63][n0..+63]; pack k-pairs -> ds_store_b32
    {
      const int kp  = t >> 2;            // k-pair 0..31 -> rows 2kp, 2kp+1
      const int n16 = (t & 3) * 16;      // 16 n columns
      float r0[16], r1[16];
#pragma unroll
      for (int c = 0; c < 4; ++c) {
        const float4 a = *(const float4*)(W + (size_t)(k0 + 2 * kp) * HD + n0 + n16 + c * 4);
        const float4 b = *(const float4*)(W + (size_t)(k0 + 2 * kp + 1) * HD + n0 + n16 + c * 4);
        r0[c * 4 + 0] = a.x; r0[c * 4 + 1] = a.y; r0[c * 4 + 2] = a.z; r0[c * 4 + 3] = a.w;
        r1[c * 4 + 0] = b.x; r1[c * 4 + 1] = b.y; r1[c * 4 + 2] = b.z; r1[c * 4 + 3] = b.w;
      }
#pragma unroll
      for (int j = 0; j < 16; ++j)
        *(unsigned*)&Bt[n16 + j][2 * kp] = pack2f(r0[j], r1[j]);
    }
    // prefetch next K tile while this one is consumed
    if (k0 + 64 < D_) {
      __builtin_prefetch(X + (size_t)(m0 + ((t * 8) >> 4)) * D_ + k0 + 64 + ((t * 8) & 15) * 4, 0, 0);
      __builtin_prefetch(W + (size_t)(k0 + 64 + (t >> 2) * 2) * HD + n0 + (t & 3) * 16, 0, 0);
    }
    __syncthreads();

#pragma unroll
    for (int kk = 0; kk < 2; ++kk) {
      const int kb = kk * 32;
      v16bf a[2];
#pragma unroll
      for (int i = 0; i < 2; ++i) {
        const int rb = wr * 32 + i * 16 + l16;
        a[i] = cat8(*(const v8bf*)&As[rb][kb + half * 16],
                    *(const v8bf*)&As[rb][kb + half * 16 + 8]);
      }
#pragma unroll
      for (int j = 0; j < 2; ++j) {
        const int cb = wc * 32 + j * 16 + l16;
        const v16bf b = *(const v16bf*)&Bt[cb][kb + half * 16];
        acc[0][j] = wmma_bf16(a[0], b, acc[0][j]);
        acc[1][j] = wmma_bf16(a[1], b, acc[1][j]);
      }
    }
  }

  // epilogue: + bias, store to per-head layout [B,H,S,DH] bf16
#pragma unroll
  for (int i = 0; i < 2; ++i) {
#pragma unroll
    for (int j = 0; j < 2; ++j) {
      const int c  = n0 + wc * 32 + j * 16 + l16;
      const float bvv = bias[c];
      const int hh = c >> 6, dh = c & 63;
#pragma unroll
      for (int r = 0; r < 8; ++r) {
        const int m = m0 + wr * 32 + i * 16 + r + half * 8;
        const int bb = m >> 10, ss = m & 1023;
        Out[(((size_t)bb * H_ + hh) * S_ + ss) * DH_ + dh] =
            (__bf16)(acc[i][j][r] + bvv);
      }
    }
  }
}

// ---------------------------------------------------------------------------
// Flash attention per head: each wave owns 16 queries, loops 64-key blocks.
// 16 WMMA / wave / iter (8 QK^T + 8 PV).
// ---------------------------------------------------------------------------
__global__ __launch_bounds__(128)
void attn_kernel(const __bf16* __restrict__ Qh, const __bf16* __restrict__ Kh,
                 const __bf16* __restrict__ Vh, __bf16* __restrict__ Ctx) {
  __shared__ __align__(32) __bf16 Ks[64][64];       // keys, row-major (key, dh)
  __shared__ __align__(32) __bf16 Vt[64][64];       // V transposed (dh, key)
  __shared__ __align__(32) __bf16 Pls[4][16][64];   // per-wave P transpose

  const int t    = threadIdx.x;
  const int wv   = t >> 5;
  const int lane = t & 31;
  const int half = lane >> 4;
  const int l16  = lane & 15;

  const int bh = blockIdx.x >> 4;               // b*H + h
  const int q0 = (blockIdx.x & 15) * 64 + wv * 16;

  // Q A-fragments for dh chunks {0,32}
  const __bf16* qptr = Qh + ((size_t)bh * S_ + q0) * DH_;
  v16bf qa[2];
#pragma unroll
  for (int i = 0; i < 2; ++i) {
    const __bf16* rp = qptr + (size_t)l16 * DH_ + i * 32;
    qa[i] = cat8(*(const v8bf*)(rp + half * 16),
                 *(const v8bf*)(rp + half * 16 + 8));
  }

  float mrow[8], lrow[8];
#pragma unroll
  for (int r = 0; r < 8; ++r) { mrow[r] = -3.0e38f; lrow[r] = 0.0f; }
  v8f Oacc[4] = {{}, {}, {}, {}};

  const __bf16* kbase = Kh + (size_t)bh * S_ * DH_;
  const __bf16* vbase = Vh + (size_t)bh * S_ * DH_;

  for (int kb = 0; kb < S_ / 64; ++kb) {
    __syncthreads();
    {  // stage K block (64 keys x 64 dh), contiguous b128 traffic
      const int row = t >> 1, coloff = (t & 1) * 32;
      const __bf16* ks = kbase + (size_t)(kb * 64 + row) * DH_ + coloff;
#pragma unroll
      for (int c = 0; c < 4; ++c)
        *(uint4*)&Ks[row][coloff + c * 8] = *(const uint4*)(ks + c * 8);
    }
    {  // stage V transposed (dh, key); pack key-pairs -> ds_store_b32
      const int rp2 = t >> 2;            // key pair 0..31
      const int dh8 = (t & 3) * 16;      // 16 dh values
      const __bf16* v0p = vbase + (size_t)(kb * 64 + 2 * rp2) * DH_ + dh8;
      const __bf16* v1p = v0p + DH_;
      const v8bf a0 = *(const v8bf*)(v0p), a1 = *(const v8bf*)(v0p + 8);
      const v8bf b0 = *(const v8bf*)(v1p), b1 = *(const v8bf*)(v1p + 8);
#pragma unroll
      for (int j = 0; j < 8; ++j) {
        *(unsigned*)&Vt[dh8 + j][2 * rp2]     = pack2b(a0[j], b0[j]);
        *(unsigned*)&Vt[dh8 + 8 + j][2 * rp2] = pack2b(a1[j], b1[j]);
      }
    }
    if (kb + 1 < S_ / 64) {
      __builtin_prefetch(kbase + (size_t)((kb + 1) * 64 + (t >> 1)) * DH_ + (t & 1) * 32, 0, 0);
      __builtin_prefetch(vbase + (size_t)((kb + 1) * 64 + (t >> 1)) * DH_ + (t & 1) * 32, 0, 0);
    }
    __syncthreads();

    // scores for four 16-key subtiles: S = Q(16x64) . K^T(64x16)
    v8f sc4[4] = {{}, {}, {}, {}};
#pragma unroll
    for (int st = 0; st < 4; ++st)
#pragma unroll
      for (int i = 0; i < 2; ++i)
        sc4[st] = wmma_bf16(qa[i],
                            *(const v16bf*)&Ks[st * 16 + l16][i * 32 + half * 16],
                            sc4[st]);

    // online softmax (row = r + half*8, 16-lane col groups)
#pragma unroll
    for (int r = 0; r < 8; ++r) {
      float v0 = sc4[0][r] * 0.125f;   // 1/sqrt(64)
      float v1 = sc4[1][r] * 0.125f;
      float v2 = sc4[2][r] * 0.125f;
      float v3 = sc4[3][r] * 0.125f;
      float mx = fmaxf(fmaxf(v0, v1), fmaxf(v2, v3));
#pragma unroll
      for (int off = 1; off < 16; off <<= 1)
        mx = fmaxf(mx, __shfl_xor(mx, off, 32));
      const float mnew = fmaxf(mrow[r], mx);
      const float scl = __expf(mrow[r] - mnew);
      const float p0 = __expf(v0 - mnew);
      const float p1 = __expf(v1 - mnew);
      const float p2 = __expf(v2 - mnew);
      const float p3 = __expf(v3 - mnew);
      float rs = (p0 + p1) + (p2 + p3);
#pragma unroll
      for (int off = 1; off < 16; off <<= 1)
        rs += __shfl_xor(rs, off, 32);
      lrow[r] = lrow[r] * scl + rs;
      mrow[r] = mnew;
#pragma unroll
      for (int j = 0; j < 4; ++j) Oacc[j][r] *= scl;
      const int prow = r + half * 8;
      Pls[wv][prow][l16]      = (__bf16)p0;
      Pls[wv][prow][16 + l16] = (__bf16)p1;
      Pls[wv][prow][32 + l16] = (__bf16)p2;
      Pls[wv][prow][48 + l16] = (__bf16)p3;
    }

    // O += P(16x64) . V(64x64): two 32-key A-frags x four 16-dh tiles
#pragma unroll
    for (int kk = 0; kk < 2; ++kk) {
      const v16bf pa = cat8(*(const v8bf*)&Pls[wv][l16][kk * 32 + half * 16],
                            *(const v8bf*)&Pls[wv][l16][kk * 32 + half * 16 + 8]);
#pragma unroll
      for (int j = 0; j < 4; ++j) {
        const v16bf bv = *(const v16bf*)&Vt[j * 16 + l16][kk * 32 + half * 16];
        Oacc[j] = wmma_bf16(pa, bv, Oacc[j]);
      }
    }
  }

  // finalize: O /= l, store bf16 ctx [bh, q, dh]
#pragma unroll
  for (int r = 0; r < 8; ++r) {
    const float inv = 1.0f / lrow[r];
    const int q = q0 + r + half * 8;
    __bf16* op = Ctx + ((size_t)bh * S_ + q) * DH_;
#pragma unroll
    for (int j = 0; j < 4; ++j)
      op[j * 16 + l16] = (__bf16)(Oacc[j][r] * inv);
  }
}

// ---------------------------------------------------------------------------
// Out = GELU(ctx @ wo + bo) + residual   (fp32 into d_out, LN done after)
// ---------------------------------------------------------------------------
__global__ __launch_bounds__(128)
void out_proj_kernel(const __bf16* __restrict__ Ctx, const float* __restrict__ W,
                     const float* __restrict__ bias, const float* __restrict__ Res,
                     float* __restrict__ Out) {
  __shared__ __align__(32) __bf16 As[64][64];
  __shared__ __align__(32) __bf16 Bt[64][64];

  const int t    = threadIdx.x;
  const int wv   = t >> 5;
  const int lane = t & 31;
  const int half = lane >> 4;
  const int l16  = lane & 15;
  const int wr   = wv >> 1, wc = wv & 1;

  const int m0 = (blockIdx.x >> 4) * 64;
  const int n0 = (blockIdx.x & 15) * 64;

  v8f acc[2][2] = {{{}, {}}, {{}, {}}};

  for (int k0 = 0; k0 < HD; k0 += 64) {
    __syncthreads();
    {  // stage A from bf16 ctx in per-head layout (32-col chunk within one head)
      const int row = t >> 1, coloff = (t & 1) * 32;
      const int m = m0 + row;
      const int bb = m >> 10, ss = m & 1023;
      const int c = k0 + coloff;
      const int hh = c >> 6, dh = c & 63;
      const __bf16* src = Ctx + (((size_t)bb * H_ + hh) * S_ + ss) * DH_ + dh;
#pragma unroll
      for (int cc = 0; cc < 4; ++cc)
        *(uint4*)&As[row][coloff + cc * 8] = *(const uint4*)(src + cc * 8);
    }
    {  // stage wo transposed, packed k-pairs
      const int kp  = t >> 2;
      const int n16 = (t & 3) * 16;
      float r0[16], r1[16];
#pragma unroll
      for (int c = 0; c < 4; ++c) {
        const float4 a = *(const float4*)(W + (size_t)(k0 + 2 * kp) * D_ + n0 + n16 + c * 4);
        const float4 b = *(const float4*)(W + (size_t)(k0 + 2 * kp + 1) * D_ + n0 + n16 + c * 4);
        r0[c * 4 + 0] = a.x; r0[c * 4 + 1] = a.y; r0[c * 4 + 2] = a.z; r0[c * 4 + 3] = a.w;
        r1[c * 4 + 0] = b.x; r1[c * 4 + 1] = b.y; r1[c * 4 + 2] = b.z; r1[c * 4 + 3] = b.w;
      }
#pragma unroll
      for (int j = 0; j < 16; ++j)
        *(unsigned*)&Bt[n16 + j][2 * kp] = pack2f(r0[j], r1[j]);
    }
    if (k0 + 64 < HD)
      __builtin_prefetch(W + (size_t)(k0 + 64 + (t >> 2) * 2) * D_ + n0 + (t & 3) * 16, 0, 0);
    __syncthreads();

#pragma unroll
    for (int kk = 0; kk < 2; ++kk) {
      const int kb = kk * 32;
      v16bf a[2];
#pragma unroll
      for (int i = 0; i < 2; ++i) {
        const int rb = wr * 32 + i * 16 + l16;
        a[i] = cat8(*(const v8bf*)&As[rb][kb + half * 16],
                    *(const v8bf*)&As[rb][kb + half * 16 + 8]);
      }
#pragma unroll
      for (int j = 0; j < 2; ++j) {
        const int cb = wc * 32 + j * 16 + l16;
        const v16bf b = *(const v16bf*)&Bt[cb][kb + half * 16];
        acc[0][j] = wmma_bf16(a[0], b, acc[0][j]);
        acc[1][j] = wmma_bf16(a[1], b, acc[1][j]);
      }
    }
  }

#pragma unroll
  for (int i = 0; i < 2; ++i) {
#pragma unroll
    for (int j = 0; j < 2; ++j) {
      const int c = n0 + wc * 32 + j * 16 + l16;
      const float bvv = bias[c];
#pragma unroll
      for (int r = 0; r < 8; ++r) {
        const int m = m0 + wr * 32 + i * 16 + r + half * 8;
        const float x = acc[i][j][r] + bvv;
        const float g = 0.5f * x * (1.0f + erff(x * 0.70710678118654752f));
        Out[(size_t)m * D_ + c] = g + Res[(size_t)m * D_ + c];
      }
    }
  }
}

// ---------------------------------------------------------------------------
// In-place LayerNorm over rows of d_out (one block per row, 256 thr x 4 elems)
// ---------------------------------------------------------------------------
__global__ __launch_bounds__(256)
void layernorm_kernel(float* __restrict__ Out, const float* __restrict__ gamma,
                      const float* __restrict__ beta) {
  __shared__ float red[2][8];
  const int t = threadIdx.x;
  float* rp = Out + (size_t)blockIdx.x * D_;
  const float4 v = *(const float4*)(rp + t * 4);
  float s = v.x + v.y + v.z + v.w;
  float q = v.x * v.x + v.y * v.y + v.z * v.z + v.w * v.w;
#pragma unroll
  for (int off = 1; off < 32; off <<= 1) {
    s += __shfl_xor(s, off, 32);
    q += __shfl_xor(q, off, 32);
  }
  if ((t & 31) == 0) { red[0][t >> 5] = s; red[1][t >> 5] = q; }
  __syncthreads();
  float ts = 0.f, tq = 0.f;
#pragma unroll
  for (int i = 0; i < 8; ++i) { ts += red[0][i]; tq += red[1][i]; }
  const float mu = ts * (1.0f / D_);
  const float rstd = rsqrtf(tq * (1.0f / D_) - mu * mu + 1e-5f);
  const float4 g = *(const float4*)(gamma + t * 4);
  const float4 bt = *(const float4*)(beta + t * 4);
  float4 o;
  o.x = (v.x - mu) * rstd * g.x + bt.x;
  o.y = (v.y - mu) * rstd * g.y + bt.y;
  o.z = (v.z - mu) * rstd * g.z + bt.z;
  o.w = (v.w - mu) * rstd * g.w + bt.w;
  *(float4*)(rp + t * 4) = o;
}

extern "C" void kernel_launch(void* const* d_in, const int* in_sizes, int n_in,
                              void* d_out, int out_size, void* d_ws, size_t ws_size,
                              hipStream_t stream) {
  (void)in_sizes; (void)n_in; (void)out_size; (void)ws_size;
  const float* q     = (const float*)d_in[0];
  const float* k     = (const float*)d_in[1];
  const float* v     = (const float*)d_in[2];
  const float* wq    = (const float*)d_in[3];
  const float* bq    = (const float*)d_in[4];
  const float* wk    = (const float*)d_in[5];
  const float* bk    = (const float*)d_in[6];
  const float* wv    = (const float*)d_in[7];
  const float* bv    = (const float*)d_in[8];
  const float* wo    = (const float*)d_in[9];
  const float* bo    = (const float*)d_in[10];
  const float* gamma = (const float*)d_in[11];
  const float* beta  = (const float*)d_in[12];
  float* out = (float*)d_out;

  char* ws = (char*)d_ws;
  const size_t seg = (size_t)NT * HD * sizeof(__bf16);  // 16 MB each
  __bf16* qh  = (__bf16*)(ws);
  __bf16* kh  = (__bf16*)(ws + seg);
  __bf16* vh  = (__bf16*)(ws + 2 * seg);
  __bf16* ctx = (__bf16*)(ws + 3 * seg);

  const dim3 blk(128);
  qkv_proj_kernel<<<dim3(2048), blk, 0, stream>>>(q, wq, bq, qh);
  qkv_proj_kernel<<<dim3(2048), blk, 0, stream>>>(k, wk, bk, kh);
  qkv_proj_kernel<<<dim3(2048), blk, 0, stream>>>(v, wv, bv, vh);
  attn_kernel<<<dim3(2048), blk, 0, stream>>>(qh, kh, vh, ctx);
  out_proj_kernel<<<dim3(2048), blk, 0, stream>>>(ctx, wo, bo, q, out);
  layernorm_kernel<<<dim3(NT), dim3(256), 0, stream>>>(out, gamma, beta);
}
